// PNA_23742579212606
// MI455X (gfx1250) — matrix operand order repaired and checked by reading, not compiled, not run on hardware
//
#include <hip/hip_runtime.h>

#define N_NODES 50000
#define N_EDGES 800000
#define HID 128
#define CLS 64
#define DELTA 3.0f

typedef __attribute__((ext_vector_type(16))) __bf16 v16bf;
typedef __attribute__((ext_vector_type(8)))  float  v8f;

union Frag16 {
  v16bf v;
  uint4 q[2];
  unsigned short s[16];
};

__device__ __forceinline__ unsigned short f2bf(float f) {
  union { float f; unsigned u; } t; t.f = f;
  unsigned r = t.u + 0x7fffu + ((t.u >> 16) & 1u);
  return (unsigned short)(r >> 16);
}

__device__ __forceinline__ void store4bf(unsigned short* p, float4 v) {
  union { unsigned short s[4]; unsigned long long u; } t;
  t.s[0] = f2bf(v.x); t.s[1] = f2bf(v.y); t.s[2] = f2bf(v.z); t.s[3] = f2bf(v.w);
  *(unsigned long long*)p = t.u;
}

__device__ __forceinline__ float4 add4(float4 a, float4 b) {
  return make_float4(a.x + b.x, a.y + b.y, a.z + b.z, a.w + b.w);
}
__device__ __forceinline__ float4 min4(float4 a, float4 b) {
  return make_float4(fminf(a.x, b.x), fminf(a.y, b.y), fminf(a.z, b.z), fminf(a.w, b.w));
}
__device__ __forceinline__ float4 max4(float4 a, float4 b) {
  return make_float4(fmaxf(a.x, b.x), fmaxf(a.y, b.y), fmaxf(a.z, b.z), fmaxf(a.w, b.w));
}
__device__ __forceinline__ float4 scl4(float4 a, float s) {
  return make_float4(a.x * s, a.y * s, a.z * s, a.w * s);
}

// ---------------- small utility kernels ----------------

__global__ void k_zero_i32(int* __restrict__ p, int n) {
  int i = blockIdx.x * blockDim.x + threadIdx.x;
  if (i < n) p[i] = 0;
}
__global__ void k_copy_i32(const int* __restrict__ a, int* __restrict__ b, int n) {
  int i = blockIdx.x * blockDim.x + threadIdx.x;
  if (i < n) b[i] = a[i];
}
__global__ void k_count(const int* __restrict__ dst, int* __restrict__ cnt, int n) {
  int i = blockIdx.x * blockDim.x + threadIdx.x;
  if (i < n) atomicAdd(&cnt[dst[i]], 1);
}
__global__ void k_scatter(const int* __restrict__ dst, int* __restrict__ cursor,
                          int* __restrict__ eidx, int n) {
  int i = blockIdx.x * blockDim.x + threadIdx.x;
  if (i < n) {
    int pos = atomicAdd(&cursor[dst[i]], 1);
    eidx[pos] = i;
  }
}
__global__ void k_cvt_bf16(const float* __restrict__ in,
                           unsigned short* __restrict__ out, int n) {
  int i = blockIdx.x * blockDim.x + threadIdx.x;
  if (i < n) out[i] = f2bf(in[i]);
}
// W[K x N] f32 (with row offset into a taller matrix) -> Wt[N][K] bf16
__global__ void k_transpose_bf16(const float* __restrict__ W,
                                 unsigned short* __restrict__ Wt,
                                 int K, int N, int rowOffset) {
  int t = blockIdx.x * blockDim.x + threadIdx.x;
  if (t >= K * N) return;
  int k = t / N, n = t - k * N;
  Wt[(size_t)n * K + k] = f2bf(W[(size_t)(rowOffset + k) * N + n]);
}

// ---------------- prefix-sum (rowptr) ----------------

__global__ void k_partial(const int* __restrict__ cnt, int* __restrict__ part, int n) {
  __shared__ int sh[256];
  int i = blockIdx.x * 256 + threadIdx.x;
  sh[threadIdx.x] = (i < n) ? cnt[i] : 0;
  __syncthreads();
  for (int s = 128; s > 0; s >>= 1) {
    if ((int)threadIdx.x < s) sh[threadIdx.x] += sh[threadIdx.x + s];
    __syncthreads();
  }
  if (threadIdx.x == 0) part[blockIdx.x] = sh[0];
}
__global__ void k_scan_part(int* __restrict__ part, int nb) {
  if (blockIdx.x == 0 && threadIdx.x == 0) {
    int acc = 0;
    for (int i = 0; i < nb; ++i) { int v = part[i]; part[i] = acc; acc += v; }
  }
}
__global__ void k_scan_final(const int* __restrict__ cnt, const int* __restrict__ part,
                             int* __restrict__ rowptr, int n) {
  __shared__ int sh[256];
  int i = blockIdx.x * 256 + threadIdx.x;
  int v = (i < n) ? cnt[i] : 0;
  sh[threadIdx.x] = v;
  __syncthreads();
  for (int off = 1; off < 256; off <<= 1) {
    int t = ((int)threadIdx.x >= off) ? sh[threadIdx.x - off] : 0;
    __syncthreads();
    sh[threadIdx.x] += t;
    __syncthreads();
  }
  int excl = sh[threadIdx.x] - v + part[blockIdx.x];
  if (i < n) rowptr[i] = excl;
  if (i == n - 1) rowptr[n] = excl + v;
}

// ---------------- generic GEMM: A[M][K] bf16 @ Wt[dout][K] -> out[M][dout] f32 ----
// Block: 16 rows of A staged to LDS via async copy, 8 waves each own 16 cols.
// A (16x32 bf16): lane m = L&15; halves 0..7 -> K = k0+16*half+(0..7);
//                 halves 8..15 -> K = k0+16*half+8+(0..7).
// B (32x16 bf16): lane n = L&15; halves j -> K = k0+16*half+j (contiguous in Wt).
// C (16x16 f32):  lane L, VGPR r -> (m = 8*half+r, n = L&15).

__global__ __launch_bounds__(256) void k_gemm(
    const unsigned short* __restrict__ A,   // [M][K] bf16
    const unsigned short* __restrict__ Wt,  // [dout][K] bf16
    const float* __restrict__ bias,         // [dout] or nullptr
    float* __restrict__ out,                // [M][dout]
    int K, int dout, int doRelu)
{
  extern __shared__ __align__(16) unsigned short smem[];  // 16 x (K+8) halves
  const int lane = threadIdx.x & 31;
  const int wave = threadIdx.x >> 5;
  const int half = lane >> 4;
  const int l16  = lane & 15;
  const int mtile = blockIdx.x * 16;
  const int ldsRow = K + 8;                 // +16B pad: bank spread, keeps 16B align

  // ---- async stage A tile (16 rows x K bf16) into LDS ----
  {
    const int kchunks = K >> 3;             // 16-byte chunks per row
    const unsigned long long gA =
        (unsigned long long)(A + (size_t)mtile * K);
    const unsigned ldsBase = (unsigned)(size_t)(void*)smem;
    for (int c = threadIdx.x; c < kchunks * 16; c += 256) {
      int row = c / kchunks;
      int wi  = c - row * kchunks;
      unsigned goff = (unsigned)((row * K + wi * 8) * 2);
      unsigned laddr = ldsBase + (unsigned)((row * ldsRow + wi * 8) * 2);
      asm volatile("global_load_async_to_lds_b128 %0, %1, %2"
                   :: "v"(laddr), "v"(goff), "s"(gA) : "memory");
    }
    asm volatile("s_wait_asynccnt 0" ::: "memory");
  }
  __syncthreads();

  const int n0 = wave * 16;
  if (n0 >= dout) return;                   // (layer-2 update uses 4 of 8 waves)

  const unsigned short* arow = smem + (size_t)l16 * ldsRow;
  const unsigned short* wrow = Wt + (size_t)(n0 + l16) * K + 16 * half;

  v8f c = {};
#pragma unroll 4
  for (int ks = 0; ks < (K >> 5); ++ks) {
    const int k0 = ks * 32;
    Frag16 a, b;
    a.q[0] = *(const uint4*)(arow + k0 + 16 * half);       // ds_load_b128
    a.q[1] = *(const uint4*)(arow + k0 + 16 * half + 8);
    const uint4* p = (const uint4*)(wrow + k0);
    b.q[0] = p[0];
    b.q[1] = p[1];
    c = __builtin_amdgcn_wmma_f32_16x16x32_bf16(false, a.v, false, b.v,
                                                (short)0, c, false, false);
  }

  const float bv = bias ? bias[n0 + l16] : 0.0f;
#pragma unroll
  for (int r = 0; r < 8; ++r) {
    const int m = 8 * half + r;
    float val = c[r] + bv;
    if (doRelu) val = fmaxf(val, 0.0f);
    out[(size_t)(mtile + m) * dout + n0 + l16] = val;
  }
}

// ---------------- node-centric aggregation + xin build (no atomics) -----------
// msg[e] = P[src[e]] + Q[dst[e]]  (Mb folded into P).
// One wave per node; lane owns 4 features. sum/min/max over in-edges in regs,
// Q[node] folded in analytically. Writes xin = [h, agg, agg*amp, agg*att] bf16.

__global__ __launch_bounds__(256) void k_aggregate(
    const float* __restrict__ P,            // [N][128]  (h@Mw_top + Mb)
    const float* __restrict__ Q,            // [N][128]  (h@Mw_bot)
    const int* __restrict__ rowptr, const int* __restrict__ eidx,
    const int* __restrict__ src,
    const unsigned short* __restrict__ hb,  // [N][128] bf16
    unsigned short* __restrict__ xin,       // [N][1280] bf16
    int nNodes)
{
  const int wave = threadIdx.x >> 5;
  const int lane = threadIdx.x & 31;
  const int node = blockIdx.x * 8 + wave;
  if (node >= nNodes) return;
  const int j0 = lane * 4;

  const int beg = rowptr[node];
  const int end = rowptr[node + 1];
  const int cnt = end - beg;

  const float INF = __builtin_huge_valf();
  float4 s  = make_float4(0.f, 0.f, 0.f, 0.f);
  float4 mn = make_float4(INF, INF, INF, INF);
  float4 mx = make_float4(-INF, -INF, -INF, -INF);

  for (int k = beg; k < end; ++k) {
    const int e  = eidx[k];
    const int sv = src[e];
    const float4 p = *(const float4*)(P + (size_t)sv * 128 + j0);
    s  = add4(s, p);
    mn = min4(mn, p);
    mx = max4(mx, p);
  }

  const float4 qb = *(const float4*)(Q + (size_t)node * 128 + j0);
  const float fc = (float)cnt;
  // mean = (sum_P + cnt*Q)/max(cnt,1); min/max = ext_P + Q (0 if no neighbors)
  float4 mean = scl4(add4(s, scl4(qb, fc)), 1.0f / fmaxf(fc, 1.0f));
  float4 mnf, mxf;
  if (cnt > 0) { mnf = add4(mn, qb); mxf = add4(mx, qb); }
  else         { mnf = make_float4(0,0,0,0); mxf = make_float4(0,0,0,0);
                 mean = make_float4(0,0,0,0); }

  const float logd = __logf(fc + 1.0f);
  const float a = logd / DELTA;
  const float t = DELTA / fmaxf(logd, 1e-6f);

  unsigned short* row = xin + (size_t)node * 1280;
  // group 0: copy h (already bf16), 4 halves = 8 bytes
  *(unsigned long long*)(row + j0) =
      *(const unsigned long long*)(hb + (size_t)node * 128 + j0);
  store4bf(row + 128  + j0, mean);
  store4bf(row + 256  + j0, mnf);
  store4bf(row + 384  + j0, mxf);
  store4bf(row + 512  + j0, scl4(mean, a));
  store4bf(row + 640  + j0, scl4(mnf, a));
  store4bf(row + 768  + j0, scl4(mxf, a));
  store4bf(row + 896  + j0, scl4(mean, t));
  store4bf(row + 1024 + j0, scl4(mnf, t));
  store4bf(row + 1152 + j0, scl4(mxf, t));
}

// ---------------- host orchestration ----------------

extern "C" void kernel_launch(void* const* d_in, const int* in_sizes, int n_in,
                              void* d_out, int out_size, void* d_ws, size_t ws_size,
                              hipStream_t stream)
{
  (void)in_sizes; (void)n_in; (void)out_size; (void)ws_size;

  const float* x   = (const float*)d_in[0];
  const int*   src = (const int*)d_in[1];
  const int*   dst = (const int*)d_in[2];
  const float* Mw[3] = {(const float*)d_in[3], (const float*)d_in[7],  (const float*)d_in[11]};
  const float* Mb[3] = {(const float*)d_in[4], (const float*)d_in[8],  (const float*)d_in[12]};
  const float* Uw[3] = {(const float*)d_in[5], (const float*)d_in[9],  (const float*)d_in[13]};
  const float* Ub[3] = {(const float*)d_in[6], (const float*)d_in[10], (const float*)d_in[14]};
  const int douts[3] = {HID, HID, CLS};

  char* p = (char*)d_ws;
  auto carve = [&](size_t bytes) -> char* {
    char* r = p;
    p += (bytes + 255) & ~(size_t)255;
    return r;
  };
  int* cnt    = (int*)carve((size_t)N_NODES * 4);
  int* rowptr = (int*)carve((size_t)(N_NODES + 1) * 4);
  int* cursor = (int*)carve((size_t)N_NODES * 4);
  int* part   = (int*)carve((size_t)256 * 4);
  int* eidx   = (int*)carve((size_t)N_EDGES * 4);
  unsigned short* hb = (unsigned short*)carve((size_t)N_NODES * 128 * 2);
  float* P     = (float*)carve((size_t)N_NODES * 128 * 4);
  float* Q     = (float*)carve((size_t)N_NODES * 128 * 4);
  float* hnext = (float*)carve((size_t)N_NODES * 128 * 4);
  unsigned short* xin = (unsigned short*)carve((size_t)N_NODES * 1280 * 2);
  unsigned short *MwtT[3], *MwtB[3], *Uwt[3];
  for (int l = 0; l < 3; ++l) {
    MwtT[l] = (unsigned short*)carve((size_t)128 * 128 * 2);
    MwtB[l] = (unsigned short*)carve((size_t)128 * 128 * 2);
    Uwt[l]  = (unsigned short*)carve((size_t)1280 * douts[l] * 2);
  }

  const int T = 256;
  const int nbScan = (N_NODES + 255) / 256;   // 196

  // Weights: split Mw into top/bottom halves, transpose all to [n][k] bf16.
  for (int l = 0; l < 3; ++l) {
    int nM = 128 * 128;
    k_transpose_bf16<<<(nM + T - 1) / T, T, 0, stream>>>(Mw[l], MwtT[l], 128, 128, 0);
    k_transpose_bf16<<<(nM + T - 1) / T, T, 0, stream>>>(Mw[l], MwtB[l], 128, 128, 128);
    int nU = 1280 * douts[l];
    k_transpose_bf16<<<(nU + T - 1) / T, T, 0, stream>>>(Uw[l], Uwt[l], 1280, douts[l], 0);
  }

  // CSR by dst (built once; reused all layers)
  k_zero_i32<<<(N_NODES + T - 1) / T, T, 0, stream>>>(cnt, N_NODES);
  k_count<<<(N_EDGES + T - 1) / T, T, 0, stream>>>(dst, cnt, N_EDGES);
  k_partial<<<nbScan, 256, 0, stream>>>(cnt, part, N_NODES);
  k_scan_part<<<1, 1, 0, stream>>>(part, nbScan);
  k_scan_final<<<nbScan, 256, 0, stream>>>(cnt, part, rowptr, N_NODES);
  k_copy_i32<<<(N_NODES + T - 1) / T, T, 0, stream>>>(rowptr, cursor, N_NODES);
  k_scatter<<<(N_EDGES + T - 1) / T, T, 0, stream>>>(dst, cursor, eidx, N_EDGES);

  const int nh = N_NODES * 128;
  k_cvt_bf16<<<(nh + T - 1) / T, T, 0, stream>>>(x, hb, nh);

  const int gTiles = N_NODES / 16;            // 3125
  for (int l = 0; l < 3; ++l) {
    // P = h @ Mw_top + Mb ; Q = h @ Mw_bot     (K=128, LDS tile 16x136 halves)
    size_t smallLds = (size_t)16 * (128 + 8) * 2;
    k_gemm<<<gTiles, 256, smallLds, stream>>>(hb, MwtT[l], Mb[l], P, 128, 128, 0);
    k_gemm<<<gTiles, 256, smallLds, stream>>>(hb, MwtB[l], nullptr, Q, 128, 128, 0);

    // Node-centric PNA aggregation + xin build (no atomics)
    k_aggregate<<<(N_NODES + 7) / 8, 256, 0, stream>>>(
        P, Q, rowptr, eidx, src, hb, xin, N_NODES);

    // Update GEMM: xin[N][1280] @ Uw -> h_next / d_out  (LDS tile 16x1288 halves)
    float* outl = (l < 2) ? hnext : (float*)d_out;
    size_t bigLds = (size_t)16 * (1280 + 8) * 2;
    k_gemm<<<gTiles, 256, bigLds, stream>>>(
        xin, Uwt[l], Ub[l], outl, 1280, douts[l], (l < 2) ? 1 : 0);

    if (l < 2) k_cvt_bf16<<<(nh + T - 1) / T, T, 0, stream>>>(hnext, hb, nh);
  }
}
